// UnifiedAGaLiTeLayer_78099685310534
// MI455X (gfx1250) — matrix operand order repaired and checked by reading, not compiled, not run on hardware
//
#include <hip/hip_runtime.h>
#include <math.h>

// ---------------------------------------------------------------------------
// UnifiedAGaLiTeLayer for MI455X (gfx1250).
// Compute-bound (~56 GFLOP of GEMM vs ~6us of HBM traffic @23.3TB/s) =>
// all dense math via v_wmma_f32_16x16x32_bf16 (fp32 accum).
// f32->bf16 conversion is done ONCE per element (staging kernels / fused
// epilogues); GEMM inner loop is pure b128 loads + WMMA at a 2:1 ratio
// (32x32 per-wave tile: A and B fragments each reused twice).
// 192MB L2 holds every GEMM operand set => stream from global (L2-resident),
// no LDS tiling, EXEC stays all-ones for WMMA.
// ---------------------------------------------------------------------------

#define T_STEPS 128
#define BATCH   32
#define ROWS    4096          // T*B
#define DMODEL  512
#define NPROJ   2656
#define NH      8
#define HD      64
#define ETA     4
#define DK      256           // ETA*HD
#define RR      8
#define DFF     2048
#define NGATE   1536          // 3*DMODEL

typedef __attribute__((ext_vector_type(16))) __bf16 v16bf;
typedef __attribute__((ext_vector_type(8)))  float  v8f;

union BF16Frag { v16bf v; uint4 q[2]; };

__device__ __forceinline__ unsigned short f2bf(float f) {
  union { float f; unsigned u; } x; x.f = f;
  unsigned r = x.u + 0x7FFFu + ((x.u >> 16) & 1u);   // round-to-nearest-even
  return (unsigned short)(r >> 16);
}

__device__ __forceinline__ float sigf(float x) { return 1.0f / (1.0f + expf(-x)); }

// ---------------------------------------------------------------------------
// C[M,N] = act( A[M,K](bf16) * W[N,K](bf16)^T [+ bias] ), fp32 accum.
// One wave = 32x32 output (2 M-tiles x 2 N-tiles); per k-chunk: 4 b128 A loads
// + 4 b128 B loads feed 4 WMMAs (2 loads per WMMA, each fragment used twice).
// 8 waves/block cover 256 rows. K compile-time -> unrolled (8 WMMAs/body).
// Fragment k order used here (contiguous, not the ISA table order; A and B match):
//   A lane: elems 0..15 -> K=hi*16+i (contiguous 16-run)
//   B lane: elems 0..15 -> K=hi*16+i (contiguous 16-run)
// ---------------------------------------------------------------------------
template <int K, int ACT, bool OUT_BF, bool HAS_BIAS>
__global__ void wmma_gemm_bf16(const unsigned short* __restrict__ A,
                               const unsigned short* __restrict__ W,
                               const float* __restrict__ bias,
                               void* __restrict__ Cout,
                               int M, int N) {
  const int lane = threadIdx.x & 31;
  const int wid  = threadIdx.x >> 5;
  const int n0 = blockIdx.x * 32;
  const int m0 = (blockIdx.y * 8 + wid) * 32;
  const int lo = lane & 15;
  const int hi = lane >> 4;

  const unsigned short* a0row = A + (size_t)(m0 + lo) * K + hi * 16;
  const unsigned short* a1row = a0row + (size_t)16 * K;
  const unsigned short* b0row = W + (size_t)(n0 + lo) * K + hi * 16;
  const unsigned short* b1row = b0row + (size_t)16 * K;

  v8f acc00 = {}, acc01 = {}, acc10 = {}, acc11 = {};
#pragma unroll 2
  for (int kk = 0; kk < K; kk += 32) {
    BF16Frag a0, a1, b0, b1;
    a0.q[0] = *(const uint4*)(a0row + kk);
    a0.q[1] = *(const uint4*)(a0row + kk + 8);
    a1.q[0] = *(const uint4*)(a1row + kk);
    a1.q[1] = *(const uint4*)(a1row + kk + 8);
    b0.q[0] = *(const uint4*)(b0row + kk);
    b0.q[1] = *(const uint4*)(b0row + kk + 8);
    b1.q[0] = *(const uint4*)(b1row + kk);
    b1.q[1] = *(const uint4*)(b1row + kk + 8);
    acc00 = __builtin_amdgcn_wmma_f32_16x16x32_bf16(false, a0.v, false, b0.v,
                                                    (short)0, acc00, false, false);
    acc01 = __builtin_amdgcn_wmma_f32_16x16x32_bf16(false, a0.v, false, b1.v,
                                                    (short)0, acc01, false, false);
    acc10 = __builtin_amdgcn_wmma_f32_16x16x32_bf16(false, a1.v, false, b0.v,
                                                    (short)0, acc10, false, false);
    acc11 = __builtin_amdgcn_wmma_f32_16x16x32_bf16(false, a1.v, false, b1.v,
                                                    (short)0, acc11, false, false);
  }

  const int col0 = n0 + lo, col1 = col0 + 16;
  float bv0 = 0.0f, bv1 = 0.0f;
  if (HAS_BIAS) { bv0 = bias[col0]; bv1 = bias[col1]; }
#pragma unroll
  for (int j = 0; j < 8; ++j) {
    const size_t row0 = (size_t)(m0 + j + hi * 8);
    const size_t row1 = row0 + 16;
    float v00 = acc00[j] + bv0;
    float v01 = acc01[j] + bv1;
    float v10 = acc10[j] + bv0;
    float v11 = acc11[j] + bv1;
    if (ACT == 1) {
      v00 = fmaxf(v00, 0.0f); v01 = fmaxf(v01, 0.0f);
      v10 = fmaxf(v10, 0.0f); v11 = fmaxf(v11, 0.0f);
    }
    if (OUT_BF) {
      unsigned short* C = (unsigned short*)Cout;
      C[row0 * N + col0] = f2bf(v00);
      C[row0 * N + col1] = f2bf(v01);
      C[row1 * N + col0] = f2bf(v10);
      C[row1 * N + col1] = f2bf(v11);
    } else {
      float* C = (float*)Cout;
      C[row0 * N + col0] = v00;
      C[row0 * N + col1] = v01;
      C[row1 * N + col0] = v10;
      C[row1 * N + col1] = v11;
    }
  }
}

// ---------------------------------------------------------------------------
// Elementwise f32 -> bf16 (8 elems/thread, b128 in / b128 out).
// ---------------------------------------------------------------------------
__global__ void cvt_bf16_kernel(const float* __restrict__ x,
                                unsigned short* __restrict__ y, int n8) {
  int i = blockIdx.x * blockDim.x + threadIdx.x;
  if (i >= n8) return;
  size_t o = (size_t)i * 8;
  float4 v0 = *(const float4*)(x + o);
  float4 v1 = *(const float4*)(x + o + 4);
  uint4 p;
  p.x = (unsigned)f2bf(v0.x) | ((unsigned)f2bf(v0.y) << 16);
  p.y = (unsigned)f2bf(v0.z) | ((unsigned)f2bf(v0.w) << 16);
  p.z = (unsigned)f2bf(v1.x) | ((unsigned)f2bf(v1.y) << 16);
  p.w = (unsigned)f2bf(v1.z) | ((unsigned)f2bf(v1.w) << 16);
  *(uint4*)(y + o) = p;
}

// ---------------------------------------------------------------------------
// LayerNorm over D=512 -> bf16 (only ever feeds a GEMM). One block per row.
// ---------------------------------------------------------------------------
__device__ __forceinline__ float block_reduce_sum(float v, float* scratch) {
#pragma unroll
  for (int off = 16; off > 0; off >>= 1) v += __shfl_down(v, off, 32);
  int lane = threadIdx.x & 31, wid = threadIdx.x >> 5;
  if (lane == 0) scratch[wid] = v;
  __syncthreads();
  if (threadIdx.x == 0) {
    float r = 0.0f;
    for (int w = 0; w < 8; ++w) r += scratch[w];
    scratch[0] = r;
  }
  __syncthreads();
  float r = scratch[0];
  __syncthreads();
  return r;
}

__global__ void layernorm_bf16_kernel(const float* __restrict__ x,
                                      const float* __restrict__ g,
                                      const float* __restrict__ b,
                                      unsigned short* __restrict__ y) {
  __shared__ float scratch[8];
  const size_t row = blockIdx.x;
  const float* xr = x + row * DMODEL;
  const int c0 = threadIdx.x, c1 = threadIdx.x + 256;
  float v0 = xr[c0], v1 = xr[c1];
  float mean = block_reduce_sum(v0 + v1, scratch) * (1.0f / DMODEL);
  float d0 = v0 - mean, d1 = v1 - mean;
  float var = block_reduce_sum(d0 * d0 + d1 * d1, scratch) * (1.0f / DMODEL);
  float rstd = rsqrtf(var + 1e-5f);
  unsigned short* yr = y + row * DMODEL;
  yr[c0] = f2bf(d0 * rstd * g[c0] + b[c0]);
  yr[c1] = f2bf(d1 * rstd * g[c1] + b[c1]);
}

// ---------------------------------------------------------------------------
// Recurrent scan + attention readout. One block per (b,h); thread d in [0,256)
// owns channel d = hd*ETA + e. All recurrent state (s, 8 k-states, 8 v-states)
// lives in registers; per step only 9 cross-lane reductions (kdq[0..7], norm)
// via wave32 shuffles + 8-wave LDS combine. attn emitted in bf16 (feeds GEMM).
// ---------------------------------------------------------------------------
__global__ void agalite_scan_kernel(const float* __restrict__ ap,      // (4096,2656)
                                    const int*   __restrict__ term,    // (T,B)
                                    const float* __restrict__ tilde_k, // (B,R,H,256)
                                    const float* __restrict__ tilde_v, // (B,R,H,64)
                                    const float* __restrict__ s_prev,  // (B,H,256)
                                    const float* __restrict__ tick,    // (B,1)
                                    unsigned short* __restrict__ attn) // (4096,512) bf16
{
  const int h = blockIdx.x & (NH - 1);
  const int b = blockIdx.x >> 3;
  const int d = threadIdx.x;          // 0..255
  const int hd = d >> 2, e = d & 3;

  __shared__ float partial[8 * 9];
  __shared__ float bc[9];

  float s = s_prev[((size_t)b * NH + h) * DK + d];
  float kst[RR], vst[RR];
#pragma unroll
  for (int r = 0; r < RR; ++r)
    kst[r] = tilde_k[(((size_t)b * RR + r) * NH + h) * DK + d];
  if (d < HD) {
#pragma unroll
    for (int r = 0; r < RR; ++r)
      vst[r] = tilde_v[(((size_t)b * RR + r) * NH + h) * HD + d];
  } else {
#pragma unroll
    for (int r = 0; r < RR; ++r) vst[r] = 0.0f;
  }

  const float tk = tick[b];
  float omega[RR];
#pragma unroll
  for (int r = 0; r < RR; ++r)
    omega[r] = -3.14159265358979323846f + (float)r * (2.0f * 3.14159265358979323846f / 7.0f);

  for (int t = 0; t < T_STEPS; ++t) {
    const size_t row = (size_t)t * BATCH + b;
    const float* pr = ap + row * NPROJ;
    const float* ph = pr + h * 320;                 // h*(5*HD)
    const float* pp = pr + (NH * HD * 5) + h * 12;  // p123 block

    float key = ph[hd];
    float qry = ph[64 + hd];
    float gam = ph[256 + hd];
    float p1 = pp[e], p2 = pp[4 + e], p3 = pp[8 + e];

    float kg  = fmaxf(key, 0.0f) * fmaxf(p1, 0.0f);   // keys_exp
    float q   = fmaxf(qry, 0.0f) * fmaxf(p2, 0.0f);   // queries_exp
    float ge  = sigf(gam) * sigf(p3);                 // gammas_exp
    float kgt = kg * ge;                              // keys_gated
    float tm  = 1.0f - (float)term[(size_t)t * BATCH + b];
    float dg  = (1.0f - ge) * tm;                     // discount_gamma

    s = dg * s + kgt;                                 // final_s step

    float tt = tk + (float)(t + 1);
    float oscv[RR];
    float vals9[9];
#pragma unroll
    for (int r = 0; r < RR; ++r) {
      oscv[r] = cosf(tt * omega[r]);
      kst[r] = dg * kst[r] + kgt * oscv[r];           // final_keys step
      vals9[r] = kst[r] * q;                          // kdq partial
    }
    vals9[8] = s * q;                                 // norm partial

#pragma unroll
    for (int i = 0; i < 9; ++i) {
#pragma unroll
      for (int off = 16; off > 0; off >>= 1)
        vals9[i] += __shfl_down(vals9[i], off, 32);
    }
    const int lane = d & 31, wid = d >> 5;
    if (lane == 0) {
#pragma unroll
      for (int i = 0; i < 9; ++i) partial[wid * 9 + i] = vals9[i];
    }
    __syncthreads();
    if (d < 9) {
      float acc = 0.0f;
      for (int w = 0; w < 8; ++w) acc += partial[w * 9 + d];
      bc[d] = acc;
    }
    __syncthreads();

    if (d < HD) {
      float bv = sigf(ph[192 + d]);                   // beta
      float vg = ph[128 + d] * bv;                    // values_gated
      float db = (1.0f - bv) * tm;                    // discount_beta
      float kv = 0.0f;
#pragma unroll
      for (int r = 0; r < RR; ++r) {
        vst[r] = db * vst[r] + vg * oscv[r];          // final_values step
        kv += vst[r] * bc[r];
      }
      float nrm = bc[8] + 1e-6f;
      attn[row * DMODEL + h * HD + d] = f2bf(kv / (16.0f * nrm));  // 2*R = 16
    }
    __syncthreads();
  }
}

// ---------------------------------------------------------------------------
// GRU gate stage 1: r = sigmoid(yW0 + xU0) ; rx = r*x  (bf16: feeds GEMM only)
// (r * x @ u2.T parses as (r*x) @ u2.T -> separate GEMM)
// ---------------------------------------------------------------------------
__global__ void gru_rx_kernel(const float* __restrict__ x,
                              const float* __restrict__ yW,   // (4096,1536)
                              const float* __restrict__ xU,   // (4096,1024)
                              unsigned short* __restrict__ rx) {
  size_t i = (size_t)blockIdx.x * blockDim.x + threadIdx.x;
  if (i >= (size_t)ROWS * DMODEL) return;
  size_t row = i >> 9;
  int c = (int)(i & 511);
  float r = sigf(yW[row * NGATE + c] + xU[row * 1024 + c]);
  rx[i] = f2bf(r * x[i]);
}

// GRU gate stage 2: z, h, blend -> f32.
__global__ void gru_out_kernel(const float* __restrict__ x,
                               const float* __restrict__ yW,    // (4096,1536)
                               const float* __restrict__ xU,    // (4096,1024)
                               const float* __restrict__ xU2r,  // (4096,512)
                               float* __restrict__ out) {
  size_t i = (size_t)blockIdx.x * blockDim.x + threadIdx.x;
  if (i >= (size_t)ROWS * DMODEL) return;
  size_t row = i >> 9;
  int c = (int)(i & 511);
  float z = sigf(yW[row * NGATE + 512 + c] + xU[row * 1024 + 512 + c] - 2.0f);
  float h = tanhf(yW[row * NGATE + 1024 + c] + xU2r[i]);
  float xv = x[i];
  out[i] = (1.0f - z) * xv + z * h;
}

// ---------------------------------------------------------------------------
// Host-side pipeline
// ---------------------------------------------------------------------------
static inline size_t alignup(size_t v) { return (v + 255) & ~(size_t)255; }

extern "C" void kernel_launch(void* const* d_in, const int* in_sizes, int n_in,
                              void* d_out, int out_size, void* d_ws, size_t ws_size,
                              hipStream_t stream) {
  const float* inputs  = (const float*)d_in[0];
  const int*   term    = (const int*)  d_in[1];
  const float* tilde_k = (const float*)d_in[2];
  const float* tilde_v = (const float*)d_in[3];
  const float* s_prev  = (const float*)d_in[4];
  const float* tick    = (const float*)d_in[5];
  const float* w_proj  = (const float*)d_in[6];
  const float* b_proj  = (const float*)d_in[7];
  const float* w_attn  = (const float*)d_in[8];
  const float* b_attn  = (const float*)d_in[9];
  const float* ln1_g   = (const float*)d_in[10];
  const float* ln1_b   = (const float*)d_in[11];
  const float* ln2_g   = (const float*)d_in[12];
  const float* ln2_b   = (const float*)d_in[13];
  const float* gru1_w  = (const float*)d_in[14];  // (3,512,512) == (1536,512)
  const float* gru1_u  = (const float*)d_in[15];
  const float* gru2_w  = (const float*)d_in[16];
  const float* gru2_u  = (const float*)d_in[17];
  const float* ffc_w1  = (const float*)d_in[18];  // (2048,512)
  const float* ffc_b1  = (const float*)d_in[19];
  const float* ffc_w2  = (const float*)d_in[20];  // (512,2048)
  const float* ffc_b2  = (const float*)d_in[21];

  char* ws = (char*)d_ws;
  size_t off = 0;
  auto takes = [&](size_t bytes) { size_t o = off; off = alignup(off + bytes); return o; };
  const size_t SZ_D   = (size_t)ROWS * DMODEL * 4;   // f32 (T*B, 512)
  const size_t SZ_Dbf = (size_t)ROWS * DMODEL * 2;   // bf16 (T*B, 512)

  // f32 scratch
  size_t o_ap   = takes((size_t)ROWS * NPROJ * 4);
  size_t o_yW   = takes((size_t)ROWS * NGATE * 4);   // reused gate1/gate2
  size_t o_xU   = takes((size_t)ROWS * 1024 * 4);    // reused
  size_t o_xu2  = takes(SZ_D);                       // reused
  size_t o_g1   = takes(SZ_D);
  // bf16 staging: weights
  size_t o_wpb  = takes((size_t)NPROJ * DMODEL * 2);
  size_t o_wab  = takes((size_t)DMODEL * DMODEL * 2);
  size_t o_g1wb = takes((size_t)NGATE * DMODEL * 2);
  size_t o_g1ub = takes((size_t)NGATE * DMODEL * 2);
  size_t o_g2wb = takes((size_t)NGATE * DMODEL * 2);
  size_t o_g2ub = takes((size_t)NGATE * DMODEL * 2);
  size_t o_fw1b = takes((size_t)DFF * DMODEL * 2);
  size_t o_fw2b = takes((size_t)DMODEL * DFF * 2);
  // bf16 staging: activations
  size_t o_inb  = takes(SZ_Dbf);
  size_t o_xnb  = takes(SZ_Dbf);
  size_t o_atb  = takes(SZ_Dbf);
  size_t o_y1b  = takes(SZ_Dbf);
  size_t o_rxb  = takes(SZ_Dbf);                     // reused
  size_t o_g1b  = takes(SZ_Dbf);
  size_t o_finb = takes(SZ_Dbf);
  size_t o_ffhb = takes((size_t)ROWS * DFF * 2);
  size_t o_fob  = takes(SZ_Dbf);
  (void)ws_size;

  float* ap   = (float*)(ws + o_ap);
  float* yW   = (float*)(ws + o_yW);
  float* xU   = (float*)(ws + o_xU);
  float* xu2  = (float*)(ws + o_xu2);
  float* g1   = (float*)(ws + o_g1);
  unsigned short* wp_b  = (unsigned short*)(ws + o_wpb);
  unsigned short* wa_b  = (unsigned short*)(ws + o_wab);
  unsigned short* g1w_b = (unsigned short*)(ws + o_g1wb);
  unsigned short* g1u_b = (unsigned short*)(ws + o_g1ub);
  unsigned short* g2w_b = (unsigned short*)(ws + o_g2wb);
  unsigned short* g2u_b = (unsigned short*)(ws + o_g2ub);
  unsigned short* fw1_b = (unsigned short*)(ws + o_fw1b);
  unsigned short* fw2_b = (unsigned short*)(ws + o_fw2b);
  unsigned short* in_b  = (unsigned short*)(ws + o_inb);
  unsigned short* xn_b  = (unsigned short*)(ws + o_xnb);
  unsigned short* at_b  = (unsigned short*)(ws + o_atb);
  unsigned short* y1_b  = (unsigned short*)(ws + o_y1b);
  unsigned short* rx_b  = (unsigned short*)(ws + o_rxb);
  unsigned short* g1_b  = (unsigned short*)(ws + o_g1b);
  unsigned short* fin_b = (unsigned short*)(ws + o_finb);
  unsigned short* ffh_b = (unsigned short*)(ws + o_ffhb);
  unsigned short* fo_b  = (unsigned short*)(ws + o_fob);
  float* out = (float*)d_out;

  const dim3 blk(256);
  const int MY = ROWS / 256;                     // 8 waves * 32 rows / block
  const int EW = (ROWS * DMODEL + 255) / 256;
  auto cvt = [&](const float* src, unsigned short* dst, size_t n) {
    int n8 = (int)(n / 8);
    cvt_bf16_kernel<<<(n8 + 255) / 256, blk, 0, stream>>>(src, dst, n8);
  };

  // 0) per-launch bf16 staging of weights + raw inputs (each element once)
  cvt(w_proj, wp_b, (size_t)NPROJ * DMODEL);
  cvt(w_attn, wa_b, (size_t)DMODEL * DMODEL);
  cvt(gru1_w, g1w_b, (size_t)NGATE * DMODEL);
  cvt(gru1_u, g1u_b, (size_t)NGATE * DMODEL);
  cvt(gru2_w, g2w_b, (size_t)NGATE * DMODEL);
  cvt(gru2_u, g2u_b, (size_t)NGATE * DMODEL);
  cvt(ffc_w1, fw1_b, (size_t)DFF * DMODEL);
  cvt(ffc_w2, fw2_b, (size_t)DMODEL * DFF);
  cvt(inputs, in_b, (size_t)ROWS * DMODEL);

  // 1) LN1 -> bf16
  layernorm_bf16_kernel<<<ROWS, blk, 0, stream>>>(inputs, ln1_g, ln1_b, xn_b);
  // 2) all_proj (f32: consumed elementwise by the scan)
  wmma_gemm_bf16<DMODEL, 0, false, true><<<dim3(NPROJ / 32, MY), blk, 0, stream>>>(
      xn_b, wp_b, b_proj, ap, ROWS, NPROJ);
  // 3) recurrent scan -> attn (bf16)
  agalite_scan_kernel<<<BATCH * NH, blk, 0, stream>>>(ap, term, tilde_k, tilde_v,
                                                      s_prev, tick, at_b);
  // 4) y1 = relu(attn @ w_attn^T + b_attn) -> bf16
  wmma_gemm_bf16<DMODEL, 1, true, true><<<dim3(DMODEL / 32, MY), blk, 0, stream>>>(
      at_b, wa_b, b_attn, y1_b, ROWS, DMODEL);
  // 5) GRU gate 1
  wmma_gemm_bf16<DMODEL, 0, false, false><<<dim3(NGATE / 32, MY), blk, 0, stream>>>(
      y1_b, g1w_b, nullptr, yW, ROWS, NGATE);
  wmma_gemm_bf16<DMODEL, 0, false, false><<<dim3(1024 / 32, MY), blk, 0, stream>>>(
      in_b, g1u_b, nullptr, xU, ROWS, 1024);
  gru_rx_kernel<<<EW, blk, 0, stream>>>(inputs, yW, xU, rx_b);
  wmma_gemm_bf16<DMODEL, 0, false, false><<<dim3(DMODEL / 32, MY), blk, 0, stream>>>(
      rx_b, g1u_b + (size_t)1024 * DMODEL, nullptr, xu2, ROWS, DMODEL);
  gru_out_kernel<<<EW, blk, 0, stream>>>(inputs, yW, xU, xu2, g1);
  cvt(g1, g1_b, (size_t)ROWS * DMODEL);
  // 6) LN2 + FFC
  layernorm_bf16_kernel<<<ROWS, blk, 0, stream>>>(g1, ln2_g, ln2_b, fin_b);
  wmma_gemm_bf16<DMODEL, 1, true, true><<<dim3(DFF / 32, MY), blk, 0, stream>>>(
      fin_b, fw1_b, ffc_b1, ffh_b, ROWS, DFF);
  wmma_gemm_bf16<DFF, 1, true, true><<<dim3(DMODEL / 32, MY), blk, 0, stream>>>(
      ffh_b, fw2_b, ffc_b2, fo_b, ROWS, DMODEL);
  // 7) GRU gate 2 (x = g1, y = ffc_out)
  wmma_gemm_bf16<DMODEL, 0, false, false><<<dim3(NGATE / 32, MY), blk, 0, stream>>>(
      fo_b, g2w_b, nullptr, yW, ROWS, NGATE);
  wmma_gemm_bf16<DMODEL, 0, false, false><<<dim3(1024 / 32, MY), blk, 0, stream>>>(
      g1_b, g2u_b, nullptr, xU, ROWS, 1024);
  gru_rx_kernel<<<EW, blk, 0, stream>>>(g1, yW, xU, rx_b);
  wmma_gemm_bf16<DMODEL, 0, false, false><<<dim3(DMODEL / 32, MY), blk, 0, stream>>>(
      rx_b, g2u_b + (size_t)1024 * DMODEL, nullptr, xu2, ROWS, DMODEL);
  gru_out_kernel<<<EW, blk, 0, stream>>>(g1, yW, xU, xu2, out);
}
